// ContrastiveModel_19061064860124
// MI455X (gfx1250) — matrix operand order repaired and checked by reading, not compiled, not run on hardware
//
#include <hip/hip_runtime.h>
#include <hip/hip_bf16.h>
#include <math.h>

#define BS_   8192
#define DHID  1024
#define DEMB  256
#define NLAB  64

#define KC      128          // K-chunk staged in LDS per step
#define BSTRIDE (KC + 8)     // halfs; 272B row stride -> 4-bank rotation, no conflicts
#define NSEG    (128 * (KC / 8) / 256)  // async copy instructions per thread per chunk (= 8)

typedef __attribute__((ext_vector_type(16))) _Float16 v16h;
typedef __attribute__((ext_vector_type(8)))  _Float16 v8h;
typedef __attribute__((ext_vector_type(8)))  float    v8f;

// ---- gfx1250 async global->LDS copy (ASYNCcnt-tracked), via inline asm ----
// Generic pointers to __shared__ carry the LDS byte offset in their low 32 bits
// (generic = {SHARED_BASE, offset32}), which is exactly what VDST wants in GV mode.
__device__ __forceinline__ void async_copy16(const _Float16* g, _Float16* l) {
  unsigned ldsoff = (unsigned)(size_t)l;
  unsigned long long ga = (unsigned long long)(size_t)g;
  asm volatile("global_load_async_to_lds_b128 %0, %1, off"
               :: "v"(ldsoff), "v"(ga) : "memory");
}

// wait until this wave's outstanding async copies <= N (they retire in order)
template <int N>
__device__ __forceinline__ void async_wait_le() {
#if __has_builtin(__builtin_amdgcn_s_wait_asynccnt)
  __builtin_amdgcn_s_wait_asynccnt(N);
#else
  asm volatile("s_wait_asynccnt %0" :: "i"(N) : "memory");
#endif
}

// ---------------- WMMA fragment helpers (CDNA5 16x16x32 f16 layout) ----------------
// combine two contiguous 8-half chunks into a v16h fragment register set
__device__ __forceinline__ v16h load_frag_pair(const _Float16* p, int gap) {
  v8h lo = *(const v8h*)(p);
  v8h hi = *(const v8h*)(p + gap);
  v16h r;
#pragma unroll
  for (int t = 0; t < 8; ++t) { r[t] = lo[t]; r[8 + t] = hi[t]; }
  return r;
}

// A-matrix 16x32 (MxK) from global, row-major, K contiguous:
// lane<16 -> row=lane, K=kk+0..15 ; lane>=16 -> row=lane-16, K=kk+16..31
__device__ __forceinline__ v16h load_a_frag(const _Float16* base, int stride, int row,
                                            int kk, int lane) {
  int kb = kk + ((lane & 16) ? 16 : 0);
  return load_frag_pair(base + (size_t)row * stride + kb, 8);
}

// B-matrix 32x16 (KxN) from LDS-staged tile (row = output col, padded stride):
// lane<16 -> col=lane, K=kk+0..15 ; lane>=16 -> col=lane-16, K=kk+16..31
__device__ __forceinline__ v16h load_b_frag_lds(const _Float16* Bt, int brow, int kk, int lane) {
  int kb = kk + ((lane & 16) ? 16 : 0);
  return load_frag_pair(Bt + brow * BSTRIDE + kb, 8);
}

// B fragment straight from global (used by the small C-GEMM)
__device__ __forceinline__ v16h load_b_frag(const _Float16* base, int stride, int brow,
                                            int kk, int lane) {
  int kb = kk + ((lane & 16) ? 16 : 0);
  return load_frag_pair(base + (size_t)brow * stride + kb, 8);
}

// issue async staging of B rows [rowOff..+127], K in [kc, kc+KC) into Bt[128][BSTRIDE]
// (8 async instructions per thread; no wait here)
__device__ __forceinline__ void stage_issue(const _Float16* src, int stride,
                                            int rowOff, int kc, _Float16* Bt, int t) {
#pragma unroll
  for (int c = t; c < 128 * (KC / 8); c += 256) {   // 16 segs of 16B per row
    int row = c >> 4, seg = c & 15;
    async_copy16(src + (size_t)(rowOff + row) * stride + kc + seg * 8,
                 Bt + row * BSTRIDE + seg * 8);
  }
}

// ---------------- small utility kernels ----------------
__global__ __launch_bounds__(256) void zero_u32_kernel(unsigned int* p, int n) {
  int i = blockIdx.x * 256 + threadIdx.x;
  if (i < n) p[i] = 0u;
}

__global__ __launch_bounds__(256) void f32_to_f16_kernel(const float* __restrict__ x,
                                                         _Float16* __restrict__ y, int n) {
  int i = blockIdx.x * 256 + threadIdx.x;
  if (i < n) y[i] = (_Float16)x[i];
}

__global__ __launch_bounds__(256) void count_labels_kernel(const int* __restrict__ labels,
                                                           int* __restrict__ cnt) {
  int i = blockIdx.x * 256 + threadIdx.x;
  if (i < BS_) atomicAdd(&cnt[labels[i]], 1);
}

// row L2-normalize: one block per row of 256 floats; writes f32 and f16 copies
__global__ __launch_bounds__(256) void rownorm_kernel(const float* __restrict__ X,
                                                      float* __restrict__ Yf,
                                                      _Float16* __restrict__ Yh) {
  __shared__ float sd[256];
  int row = blockIdx.x, t = threadIdx.x;
  float v = X[(size_t)row * DEMB + t];
  sd[t] = v * v;
  __syncthreads();
#pragma unroll
  for (int s = 128; s > 0; s >>= 1) {
    if (t < s) sd[t] += sd[t + s];
    __syncthreads();
  }
  float nrm = fmaxf(sqrtf(sd[0]), 1e-8f);
  float y = v / nrm;
  Yf[(size_t)row * DEMB + t] = y;
  Yh[(size_t)row * DEMB + t] = (_Float16)y;
}

// ---------------- GEMM1: emb = mask @ W^T + b   (M=8192, K=1024, N=256) ----------------
__global__ __launch_bounds__(256) void gemm1_kernel(const _Float16* __restrict__ Ah,
                                                    const _Float16* __restrict__ Wh,
                                                    const float* __restrict__ bias,
                                                    float* __restrict__ emb) {
  __shared__ _Float16 Bt0[128 * BSTRIDE];
  __shared__ _Float16 Bt1[128 * BSTRIDE];
  int t = threadIdx.x, w = t >> 5, lane = t & 31;
  int rowBase = blockIdx.x * 128 + w * 16;
  int colBase = blockIdx.y * 128;
  v8f acc[8] = {};
  int arow = rowBase + (lane & 15);
  const int NC = DHID / KC;

  stage_issue(Wh, DHID, colBase, 0, Bt0, t);
  for (int c = 0; c < NC; ++c) {
    _Float16* cur = (c & 1) ? Bt1 : Bt0;
    _Float16* nxt = (c & 1) ? Bt0 : Bt1;
    if (c + 1 < NC) {                 // overlap next chunk's DMA with this chunk's math
      stage_issue(Wh, DHID, colBase, (c + 1) * KC, nxt, t);
      async_wait_le<NSEG>();          // chunk c retired (in-order completion)
    } else {
      async_wait_le<0>();
    }
    __syncthreads();
#pragma unroll
    for (int kk = 0; kk < KC; kk += 32) {
      v16h a = load_a_frag(Ah, DHID, arow, c * KC + kk, lane);
#pragma unroll
      for (int nf = 0; nf < 8; ++nf) {
        v16h b = load_b_frag_lds(cur, nf * 16 + (lane & 15), kk, lane);
        acc[nf] = __builtin_amdgcn_wmma_f32_16x16x32_f16(false, a, false, b,
                                                         (short)0, acc[nf], false, false);
      }
    }
    __syncthreads();                  // all waves done reading cur -> reusable
  }
  int mhalf = (lane & 16) ? 8 : 0;
#pragma unroll
  for (int nf = 0; nf < 8; ++nf)
#pragma unroll
    for (int r = 0; r < 8; ++r) {
      int i = rowBase + mhalf + r;
      int n = colBase + nf * 16 + (lane & 15);
      emb[(size_t)i * DEMB + n] = acc[nf][r] + bias[n];
    }
}

// ---------------- C = en @ ln^T  (M=8192, K=256, N=64) ----------------
__global__ __launch_bounds__(256) void cgemm_kernel(const _Float16* __restrict__ enh,
                                                    const _Float16* __restrict__ lnh,
                                                    float* __restrict__ Cm) {
  int w = threadIdx.x >> 5, lane = threadIdx.x & 31;
  int rowBase = blockIdx.x * 128 + w * 16;
  v8f acc[4] = {};
  int arow = rowBase + (lane & 15);
  for (int kk = 0; kk < DEMB; kk += 32) {
    v16h a = load_a_frag(enh, DEMB, arow, kk, lane);
#pragma unroll
    for (int nf = 0; nf < 4; ++nf) {
      v16h b = load_b_frag(lnh, DEMB, nf * 16 + (lane & 15), kk, lane);
      acc[nf] = __builtin_amdgcn_wmma_f32_16x16x32_f16(false, a, false, b,
                                                       (short)0, acc[nf], false, false);
    }
  }
  int mhalf = (lane & 16) ? 8 : 0;
#pragma unroll
  for (int nf = 0; nf < 4; ++nf)
#pragma unroll
    for (int r = 0; r < 8; ++r) {
      int i = rowBase + mhalf + r;
      int n = nf * 16 + (lane & 15);
      Cm[(size_t)i * NLAB + n] = acc[nf][r];
    }
}

// ---------------- Fused S kernel: N[i] += sum_{j in tile, label!=} exp(en_i . en_j) ----------------
__global__ __launch_bounds__(256) void s_fused_kernel(const _Float16* __restrict__ enh,
                                                      const int* __restrict__ labels,
                                                      float* __restrict__ Nacc) {
  __shared__ _Float16 Bt0[128 * BSTRIDE];
  __shared__ _Float16 Bt1[128 * BSTRIDE];
  __shared__ int rowLab[128], colLab[128];
  __shared__ float red[8][16][16];
  int t = threadIdx.x, w = t >> 5, lane = t & 31;
  int rowTile = blockIdx.x * 128, colTile = blockIdx.y * 128;
  if (t < 128) rowLab[t] = labels[rowTile + t];
  else         colLab[t - 128] = labels[colTile + t - 128];

  v8f acc[8] = {};
  int arow = rowTile + w * 16 + (lane & 15);
  const int NC = DEMB / KC;           // = 2

  stage_issue(enh, DEMB, colTile, 0, Bt0, t);
#pragma unroll
  for (int c = 0; c < NC; ++c) {
    _Float16* cur = (c & 1) ? Bt1 : Bt0;
    _Float16* nxt = (c & 1) ? Bt0 : Bt1;
    if (c + 1 < NC) {
      stage_issue(enh, DEMB, colTile, (c + 1) * KC, nxt, t);
      async_wait_le<NSEG>();
    } else {
      async_wait_le<0>();
    }
    __syncthreads();
#pragma unroll
    for (int kk = 0; kk < KC; kk += 32) {
      v16h a = load_a_frag(enh, DEMB, arow, c * KC + kk, lane);
#pragma unroll
      for (int nf = 0; nf < 8; ++nf) {
        v16h b = load_b_frag_lds(cur, nf * 16 + (lane & 15), kk, lane);
        acc[nf] = __builtin_amdgcn_wmma_f32_16x16x32_f16(false, a, false, b,
                                                         (short)0, acc[nf], false, false);
      }
    }
    __syncthreads();
  }

  // epilogue: exp + different-label mask, partial per (row, lane-column)
  int mhalf = (lane & 16) ? 8 : 0;
#pragma unroll
  for (int r = 0; r < 8; ++r) {
    int li = rowLab[w * 16 + mhalf + r];
    float s = 0.f;
#pragma unroll
    for (int nf = 0; nf < 8; ++nf) {
      int lj = colLab[nf * 16 + (lane & 15)];
      if (lj != li) s += __expf(acc[nf][r]);
    }
    red[w][mhalf + r][lane & 15] = s;
  }
  __syncthreads();
  if (t < 128) {
    int w2 = t >> 4, row = t & 15;
    float s = 0.f;
#pragma unroll
    for (int c = 0; c < 16; ++c) s += red[w2][row][c];
    atomicAdd(&Nacc[rowTile + w2 * 16 + row], s);
  }
}

// ---------------- inter loss: same-label terms (exact f32 recompute) + diff-label closed form ----------------
__global__ __launch_bounds__(256) void inter_kernel(const float* __restrict__ enf,
                                                    const int* __restrict__ labels,
                                                    const float* __restrict__ Nv,
                                                    const int* __restrict__ cnt,
                                                    float* __restrict__ acc) {
  __shared__ float ei[DEMB];
  __shared__ float sd[256];
  int i = blockIdx.x, t = threadIdx.x;
  ei[t] = enf[(size_t)i * DEMB + t];
  __syncthreads();
  int li = labels[i];
  float Ni = Nv[i];
  float s = 0.f;
  for (int j = t; j < BS_; j += 256) {
    if (labels[j] == li && j != i) {
      const float4* pj = (const float4*)(enf + (size_t)j * DEMB);
      float d = 0.f;
#pragma unroll 8
      for (int k = 0; k < DEMB / 4; ++k) {
        float4 q = pj[k];
        d += ei[4 * k] * q.x + ei[4 * k + 1] * q.y + ei[4 * k + 2] * q.z + ei[4 * k + 3] * q.w;
      }
      s += -d + logf(Ni + __expf(d));
    }
  }
  if (t == 0) s += (float)(BS_ - cnt[li]) * logf(Ni + 1.0f);
  sd[t] = s;
  __syncthreads();
#pragma unroll
  for (int st = 128; st > 0; st >>= 1) {
    if (t < st) sd[t] += sd[t + st];
    __syncthreads();
  }
  if (t == 0) atomicAdd(acc, sd[0]);
}

// ---------------- T[l] = sum_j exp(C[j,l]);  V[l] = sum_{label_j==l} exp(C[j,l]) ----------------
__global__ __launch_bounds__(256) void protoTV_kernel(const float* __restrict__ Cm,
                                                      const int* __restrict__ labels,
                                                      float* __restrict__ T,
                                                      float* __restrict__ V) {
  __shared__ float sdT[256], sdV[256];
  int l = blockIdx.x, t = threadIdx.x;
  float ts = 0.f, vs = 0.f;
  for (int j = t; j < BS_; j += 256) {
    float e = __expf(Cm[(size_t)j * NLAB + l]);
    ts += e;
    if (labels[j] == l) vs += e;
  }
  sdT[t] = ts; sdV[t] = vs;
  __syncthreads();
#pragma unroll
  for (int st = 128; st > 0; st >>= 1) {
    if (t < st) { sdT[t] += sdT[t + st]; sdV[t] += sdV[t + st]; }
    __syncthreads();
  }
  if (t == 0) { T[l] = sdT[0]; V[l] = sdV[0]; }
}

// ---------------- per-sample prototype terms ----------------
__global__ __launch_bounds__(256) void proto_row_kernel(const float* __restrict__ Cm,
                                                        const int* __restrict__ labels,
                                                        const float* __restrict__ T,
                                                        const float* __restrict__ V,
                                                        float* __restrict__ acc) {
  __shared__ float sd[256];
  int i = blockIdx.x * 256 + threadIdx.x;
  int li = labels[i];
  float pos = Cm[(size_t)i * NLAB + li];
  float ep = __expf(pos);
  float n1 = 0.f;
#pragma unroll
  for (int l = 0; l < NLAB; ++l)
    if (l != li) n1 += __expf(Cm[(size_t)i * NLAB + l]);
  float n2 = T[li] - V[li];
  float s = (-pos + logf(n1 + ep)) + (-pos + logf(n2 + ep));
  sd[threadIdx.x] = s;
  __syncthreads();
#pragma unroll
  for (int st = 128; st > 0; st >>= 1) {
    if (threadIdx.x < st) sd[threadIdx.x] += sd[threadIdx.x + st];
    __syncthreads();
  }
  if (threadIdx.x == 0) atomicAdd(acc, sd[0]);
}

__global__ void finalize_kernel(const float* __restrict__ interAcc,
                                const float* __restrict__ protoAcc,
                                float* __restrict__ out) {
  float inter = interAcc[0] / ((float)BS_ * (float)BS_);
  float proto = protoAcc[0] / (float)BS_;  // = mean(t1) + mean(t2)
  out[0] = 0.5f * inter + 0.5f * proto;
}

// ---------------- host launcher ----------------
extern "C" void kernel_launch(void* const* d_in, const int* in_sizes, int n_in,
                              void* d_out, int out_size, void* d_ws, size_t ws_size,
                              hipStream_t stream) {
  const float* mask_emb  = (const float*)d_in[0];   // [8192,1024]
  const float* W         = (const float*)d_in[1];   // [256,1024]
  const float* bias      = (const float*)d_in[2];   // [256]
  const float* label_emb = (const float*)d_in[3];   // [64,256]
  const int*   labels    = (const int*)d_in[4];     // [8192]

  char* base = (char*)d_ws;
  size_t off = 0;
  auto carve = [&](size_t bytes) -> char* {
    char* p = base + off;
    off = (off + bytes + 255) & ~(size_t)255;
    return p;
  };
  _Float16* mask_h = (_Float16*)carve((size_t)BS_ * DHID * 2);
  _Float16* W_h    = (_Float16*)carve((size_t)256 * DHID * 2);
  float*    emb    = (float*)   carve((size_t)BS_ * DEMB * 4);
  _Float16* en_h   = (_Float16*)carve((size_t)BS_ * DEMB * 2);
  float*    en_f   = (float*)   carve((size_t)BS_ * DEMB * 4);
  _Float16* ln_h   = (_Float16*)carve((size_t)NLAB * DEMB * 2);
  float*    ln_f   = (float*)   carve((size_t)NLAB * DEMB * 4);
  float*    Cm     = (float*)   carve((size_t)BS_ * NLAB * 4);
  // stats region (zeroed every call): N[8192] | cnt[64] | T[64] | V[64] | acc[2]
  float* stats = (float*)carve((size_t)(BS_ + NLAB * 3 + 16) * 4);
  float* Nv        = stats;
  int*   cnt       = (int*)(stats + BS_);
  float* T         = stats + BS_ + NLAB;
  float* V         = stats + BS_ + 2 * NLAB;
  float* interAcc  = stats + BS_ + 3 * NLAB;
  float* protoAcc  = interAcc + 1;

  int statsN = BS_ + 3 * NLAB + 16;
  zero_u32_kernel<<<(statsN + 255) / 256, 256, 0, stream>>>((unsigned int*)stats, statsN);

  f32_to_f16_kernel<<<(BS_ * DHID) / 256, 256, 0, stream>>>(mask_emb, mask_h, BS_ * DHID);
  f32_to_f16_kernel<<<(256 * DHID) / 256, 256, 0, stream>>>(W, W_h, 256 * DHID);
  count_labels_kernel<<<BS_ / 256, 256, 0, stream>>>(labels, cnt);

  gemm1_kernel<<<dim3(BS_ / 128, DEMB / 128), 256, 0, stream>>>(mask_h, W_h, bias, emb);

  rownorm_kernel<<<BS_, 256, 0, stream>>>(emb, en_f, en_h);
  rownorm_kernel<<<NLAB, 256, 0, stream>>>(label_emb, ln_f, ln_h);

  cgemm_kernel<<<BS_ / 128, 256, 0, stream>>>(en_h, ln_h, Cm);

  s_fused_kernel<<<dim3(BS_ / 128, BS_ / 128), 256, 0, stream>>>(en_h, labels, Nv);

  protoTV_kernel<<<NLAB, 256, 0, stream>>>(Cm, labels, T, V);
  inter_kernel<<<BS_, 256, 0, stream>>>(en_f, labels, Nv, cnt, interAcc);
  proto_row_kernel<<<BS_ / 256, 256, 0, stream>>>(Cm, labels, T, V, protoAcc);

  finalize_kernel<<<1, 1, 0, stream>>>(interAcc, protoAcc, (float*)d_out);
}
